// QuantLinear_86019605004986
// MI455X (gfx1250) — matrix operand order repaired and checked by reading, not compiled, not run on hardware
//
#include <hip/hip_runtime.h>

// GPTQ int4 linear: out = x @ dequant(qweight,qzeros,scales,g_idx) + bias
// M=8192, K=4096, N=4096. Compute-bound -> 3-stage pipeline:
//   1) cvt_x_bf16:     x fp32 -> bf16 (once)
//   2) dequant_wt:     int4 -> bf16 Wt[N][K] (once)
//   3) gemm_bf16_pipe: double-buffered LDS bf16 WMMA GEMM, async global->LDS
// Fallback to fused single kernel if workspace < 96MB.

typedef __attribute__((ext_vector_type(16))) __bf16 v16bf;
typedef __attribute__((ext_vector_type(8)))  float  v8f;

#define K_DIM 4096
#define N_DIM 4096

struct U2 { unsigned int x, y; };
struct U4 { unsigned int x, y, z, w; };
struct F4 { float x, y, z, w; };

union FragBits { U4 u[2]; v16bf v; };

// fp32 -> bf16 round-to-nearest-even
static __device__ inline unsigned short f2bf(float f) {
  union { float f; unsigned int u; } c; c.f = f;
  unsigned int r = c.u + 0x7FFFu + ((c.u >> 16) & 1u);
  return (unsigned short)(r >> 16);
}

#define ASG __attribute__((address_space(1)))
#define ASL __attribute__((address_space(3)))

// Signature (confirmed via clang diagnostic):
//   void __builtin_amdgcn_global_load_async_to_lds_b128(v4i AS1* src,
//       v4i AS3* dst, imm int offset, imm int cpol)
typedef int V4I __attribute__((vector_size(16)));

#if defined(__has_builtin)
#  if __has_builtin(__builtin_amdgcn_global_load_async_to_lds_b128) && \
      __has_builtin(__builtin_amdgcn_s_wait_asynccnt)
#    define HAVE_ASYNC_LDS 1
#  endif
#endif
#ifndef HAVE_ASYNC_LDS
#  define HAVE_ASYNC_LDS 0
#endif

// ---------------------------------------------------------------------------
// Pre-pass 1: x fp32 -> bf16 (8 elements / thread)
// ---------------------------------------------------------------------------
__global__ __launch_bounds__(256)
void cvt_x_bf16(const float* __restrict__ x, unsigned short* __restrict__ xb,
                int n8) {
  const int id = blockIdx.x * 256 + threadIdx.x;
  if (id >= n8) return;
  const F4 a = reinterpret_cast<const F4*>(x)[2 * id];
  const F4 b = reinterpret_cast<const F4*>(x)[2 * id + 1];
  U4 pk;
  pk.x = (unsigned int)f2bf(a.x) | ((unsigned int)f2bf(a.y) << 16);
  pk.y = (unsigned int)f2bf(a.z) | ((unsigned int)f2bf(a.w) << 16);
  pk.z = (unsigned int)f2bf(b.x) | ((unsigned int)f2bf(b.y) << 16);
  pk.w = (unsigned int)f2bf(b.z) | ((unsigned int)f2bf(b.w) << 16);
  reinterpret_cast<U4*>(xb)[id] = pk;
}

// ---------------------------------------------------------------------------
// Pre-pass 2: dequant int4 -> bf16, transposed Wt[N][K] (K contiguous).
// ---------------------------------------------------------------------------
__global__ __launch_bounds__(256)
void dequant_wt_bf16(const int* __restrict__ qweight,
                     const unsigned int* __restrict__ qzeros,
                     const float* __restrict__ scales,
                     const int* __restrict__ g_idx,
                     unsigned short* __restrict__ wt) {
  const int id = blockIdx.x * 256 + threadIdx.x; // 2M threads
  const int kw = id & (K_DIM / 8 - 1);           // packed k-row 0..511
  const int n  = id >> 9;                        // out feature
  const int g  = g_idx[kw * 8];
  const float s = scales[g * N_DIM + n];
  const unsigned int zw = qzeros[g * (N_DIM / 8) + (n >> 3)];
  const int z = (int)((zw >> ((n & 7) * 4)) & 0xFu) + 1; // GPTQ zero+1
  const unsigned int q = (unsigned int)qweight[(size_t)kw * N_DIM + n];
  unsigned short h[8];
#pragma unroll
  for (int j = 0; j < 8; ++j) {
    const int qv = (int)((q >> (j * 4)) & 0xFu);
    h[j] = f2bf(s * (float)(qv - z));
  }
  U4 pk;
  pk.x = (unsigned int)h[0] | ((unsigned int)h[1] << 16);
  pk.y = (unsigned int)h[2] | ((unsigned int)h[3] << 16);
  pk.z = (unsigned int)h[4] | ((unsigned int)h[5] << 16);
  pk.w = (unsigned int)h[6] | ((unsigned int)h[7] << 16);
  reinterpret_cast<U4*>(wt + (size_t)n * K_DIM)[kw] = pk;
}

// ---------------------------------------------------------------------------
// Main GEMM: C[M,N] = Xb[M,K] * Wt[N,K]^T + bias.
// 128x128 block tile, K-step 64, 8 waves (2x4), 4x2 16x16 accum tiles/wave.
// Single-body loop (#pragma unroll 1) so accumulators stay register-tied
// (no v_mov shuffles / hazard nops); unconditional wrapped prefetch avoids
// last-iteration peeling.
// ---------------------------------------------------------------------------
#define LDS_STRIDE 72   // 64 bf16 + 8 pad: 144B rows, 16B-aligned chunks
#define LDS_TILE   (128 * LDS_STRIDE)
#define NT         (K_DIM / 64)

__global__ __launch_bounds__(256)
void gemm_bf16_pipe(const unsigned short* __restrict__ A,   // [M][K] bf16
                    const unsigned short* __restrict__ Wt,  // [N][K] bf16
                    const float* __restrict__ bias,
                    float* __restrict__ out) {
  __shared__ unsigned short ldsA[2][LDS_TILE];
  __shared__ unsigned short ldsB[2][LDS_TILE];

  const int t     = threadIdx.x;
  const int lane  = t & 31;
  const int wave  = t >> 5;
  const int lm    = lane & 15;
  const int lh    = lane >> 4;
  const int waveM = wave & 1;
  const int waveN = wave >> 1;

  const int Mbase = blockIdx.y * 128;
  const int Nbase = blockIdx.x * 128;

  // Staging map: thread -> (row/col, 32-element segment); 4x16B chunks each.
  const int rowc = t >> 1;
  const int seg  = t & 1;
  const unsigned short* gA = A  + (size_t)(Mbase + rowc) * K_DIM + seg * 32;
  const unsigned short* gB = Wt + (size_t)(Nbase + rowc) * K_DIM + seg * 32;
  const int lbase = rowc * LDS_STRIDE + seg * 32;

  // Per-wave fragment base offsets (element units) within one LDS buffer.
  const int aFragBase = (waveM * 64 + lm) * LDS_STRIDE + lh * 8;
  const int bFragBase = (waveN * 32 + lm) * LDS_STRIDE + lh * 8;

  v8f acc[4][2];
#pragma unroll
  for (int mt = 0; mt < 4; ++mt)
#pragma unroll
    for (int nt = 0; nt < 2; ++nt) {
      v8f z = {0.f, 0.f, 0.f, 0.f, 0.f, 0.f, 0.f, 0.f};
      acc[mt][nt] = z;
    }

  auto compute = [&](int b) {
#pragma unroll
    for (int kk = 0; kk < 2; ++kk) {
      const int ks = kk * 32;
      FragBits fa[4], fb[2];
#pragma unroll
      for (int mt = 0; mt < 4; ++mt) {
        const int e0 = b * LDS_TILE + aFragBase + mt * (16 * LDS_STRIDE) + ks;
        fa[mt].u[0] = *reinterpret_cast<const U4*>(&ldsA[0][e0]);
        fa[mt].u[1] = *reinterpret_cast<const U4*>(&ldsA[0][e0 + 16]);
      }
#pragma unroll
      for (int nt = 0; nt < 2; ++nt) {
        const int e0 = b * LDS_TILE + bFragBase + nt * (16 * LDS_STRIDE) + ks;
        fb[nt].u[0] = *reinterpret_cast<const U4*>(&ldsB[0][e0]);
        fb[nt].u[1] = *reinterpret_cast<const U4*>(&ldsB[0][e0 + 16]);
      }
#pragma unroll
      for (int mt = 0; mt < 4; ++mt)
#pragma unroll
        for (int nt = 0; nt < 2; ++nt)
          acc[mt][nt] = __builtin_amdgcn_wmma_f32_16x16x32_bf16(
              false, fa[mt].v, false, fb[nt].v, (short)0, acc[mt][nt],
              false, false);
    }
  };

#if HAVE_ASYNC_LDS
  auto stage_async = [&](int b, int kb) {
    ASG V4I* sa = (ASG V4I*)(gA + kb);
    ASG V4I* sb = (ASG V4I*)(gB + kb);
    ASL V4I* da = (ASL V4I*)&ldsA[b][lbase];
    ASL V4I* db = (ASL V4I*)&ldsB[b][lbase];
    __builtin_amdgcn_global_load_async_to_lds_b128(sa, da,  0, 0);
    __builtin_amdgcn_global_load_async_to_lds_b128(sa, da, 16, 0);
    __builtin_amdgcn_global_load_async_to_lds_b128(sa, da, 32, 0);
    __builtin_amdgcn_global_load_async_to_lds_b128(sa, da, 48, 0);
    __builtin_amdgcn_global_load_async_to_lds_b128(sb, db,  0, 0);
    __builtin_amdgcn_global_load_async_to_lds_b128(sb, db, 16, 0);
    __builtin_amdgcn_global_load_async_to_lds_b128(sb, db, 32, 0);
    __builtin_amdgcn_global_load_async_to_lds_b128(sb, db, 48, 0);
  };

  stage_async(0, 0);
#pragma unroll 1
  for (int i = 0; i < NT; ++i) {
    const int b = i & 1;
    __builtin_amdgcn_s_wait_asynccnt(0); // tile i landed (this wave's loads)
    __syncthreads();                     // all waves landed; prior reads done
    // Unconditional prefetch (wrapped on last iter -> unused buffer, no peel).
    const int kbn = (i + 1 == NT) ? 0 : (i + 1) * 64;
    stage_async(b ^ 1, kbn);             // overlaps with compute below
    compute(b);
  }
#else
  U4 pa[4], pb[4];
  auto load_regs = [&](int kb) {
#pragma unroll
    for (int j = 0; j < 4; ++j) {
      pa[j] = *reinterpret_cast<const U4*>(gA + kb + j * 8);
      pb[j] = *reinterpret_cast<const U4*>(gB + kb + j * 8);
    }
  };
  load_regs(0);
#pragma unroll 1
  for (int i = 0; i < NT; ++i) {
    const int b = i & 1;
#pragma unroll
    for (int j = 0; j < 4; ++j) {
      *reinterpret_cast<U4*>(&ldsA[b][lbase + j * 8]) = pa[j];
      *reinterpret_cast<U4*>(&ldsB[b][lbase + j * 8]) = pb[j];
    }
    __syncthreads();
    const int kbn = (i + 1 == NT) ? 0 : (i + 1) * 64;
    load_regs(kbn); // global loads overlap compute
    compute(b);
  }
#endif

  // Epilogue: C/D layout -> row = r + 8*lh, col = lm
#pragma unroll
  for (int mt = 0; mt < 4; ++mt) {
#pragma unroll
    for (int nt = 0; nt < 2; ++nt) {
      const int row0 = Mbase + waveM * 64 + mt * 16 + lh * 8;
      const int col  = Nbase + waveN * 32 + nt * 16 + lm;
      const float bv = bias[col];
#pragma unroll
      for (int r = 0; r < 8; ++r)
        out[(size_t)(row0 + r) * N_DIM + col] = acc[mt][nt][r] + bv;
    }
  }
}

// ---------------------------------------------------------------------------
// Fallback: fused single-kernel version, used if ws too small.
// ---------------------------------------------------------------------------
#define FLDS_STRIDE 40

__global__ __launch_bounds__(256)
void gptq_gemm_fused(const float* __restrict__ x,
                     const int* __restrict__ qweight,
                     const unsigned int* __restrict__ qzeros,
                     const float* __restrict__ scales,
                     const int* __restrict__ g_idx,
                     const float* __restrict__ bias,
                     float* __restrict__ out) {
  __shared__ unsigned short ldsA[128 * FLDS_STRIDE];
  __shared__ unsigned short ldsB[128 * FLDS_STRIDE];

  const int t = threadIdx.x, lane = t & 31, wave = t >> 5;
  const int lm = lane & 15, lh = lane >> 4;
  const int waveM = wave & 1, waveN = wave >> 1;
  const int Mbase = blockIdx.y * 128, Nbase = blockIdx.x * 128;

  v8f acc[4][2];
#pragma unroll
  for (int mt = 0; mt < 4; ++mt)
#pragma unroll
    for (int nt = 0; nt < 2; ++nt) {
      v8f z = {0.f, 0.f, 0.f, 0.f, 0.f, 0.f, 0.f, 0.f};
      acc[mt][nt] = z;
    }

#pragma unroll 1
  for (int kb = 0; kb < K_DIM; kb += 32) {
    __syncthreads();
    {
      const int cg = t & 7, r0 = t >> 3;
#pragma unroll
      for (int p = 0; p < 4; ++p) {
        const int row = p * 32 + r0;
        const F4 v = *reinterpret_cast<const F4*>(
            x + (size_t)(Mbase + row) * K_DIM + kb + cg * 4);
        U2 pk;
        pk.x = (unsigned int)f2bf(v.x) | ((unsigned int)f2bf(v.y) << 16);
        pk.y = (unsigned int)f2bf(v.z) | ((unsigned int)f2bf(v.w) << 16);
        *reinterpret_cast<U2*>(&ldsA[row * FLDS_STRIDE + cg * 4]) = pk;
      }
    }
    {
      const int g = g_idx[kb];
#pragma unroll
      for (int w = 0; w < 2; ++w) {
        const int idx = t + w * 256, pr = idx >> 7, nl = idx & 127;
        const int n = Nbase + nl;
        const float s = scales[g * N_DIM + n];
        const unsigned int zw = qzeros[g * (N_DIM / 8) + (n >> 3)];
        const int z = (int)((zw >> ((n & 7) * 4)) & 0xFu) + 1;
        const unsigned int qw =
            (unsigned int)qweight[(size_t)((kb >> 3) + pr) * N_DIM + n];
        unsigned short h[8];
#pragma unroll
        for (int j = 0; j < 8; ++j) {
          const int q = (int)((qw >> (j * 4)) & 0xFu);
          h[j] = f2bf(s * (float)(q - z));
        }
        U4 pk;
        pk.x = (unsigned int)h[0] | ((unsigned int)h[1] << 16);
        pk.y = (unsigned int)h[2] | ((unsigned int)h[3] << 16);
        pk.z = (unsigned int)h[4] | ((unsigned int)h[5] << 16);
        pk.w = (unsigned int)h[6] | ((unsigned int)h[7] << 16);
        *reinterpret_cast<U4*>(&ldsB[nl * FLDS_STRIDE + pr * 8]) = pk;
      }
    }
    __syncthreads();

    FragBits fa[4], fb[2];
#pragma unroll
    for (int mt = 0; mt < 4; ++mt) {
      const int e0 = (waveM * 64 + mt * 16 + lm) * FLDS_STRIDE + lh * 8;
      fa[mt].u[0] = *reinterpret_cast<const U4*>(&ldsA[e0]);
      fa[mt].u[1] = *reinterpret_cast<const U4*>(&ldsA[e0 + 16]);
    }
#pragma unroll
    for (int nt = 0; nt < 2; ++nt) {
      const int e0 = (waveN * 32 + nt * 16 + lm) * FLDS_STRIDE + lh * 8;
      fb[nt].u[0] = *reinterpret_cast<const U4*>(&ldsB[e0]);
      fb[nt].u[1] = *reinterpret_cast<const U4*>(&ldsB[e0 + 16]);
    }
#pragma unroll
    for (int mt = 0; mt < 4; ++mt)
#pragma unroll
      for (int nt = 0; nt < 2; ++nt)
        acc[mt][nt] = __builtin_amdgcn_wmma_f32_16x16x32_bf16(
            false, fa[mt].v, false, fb[nt].v, (short)0, acc[mt][nt],
            false, false);
  }

#pragma unroll
  for (int mt = 0; mt < 4; ++mt) {
#pragma unroll
    for (int nt = 0; nt < 2; ++nt) {
      const int row0 = Mbase + waveM * 64 + mt * 16 + lh * 8;
      const int col  = Nbase + waveN * 32 + nt * 16 + lm;
      const float bv = bias[col];
#pragma unroll
      for (int r = 0; r < 8; ++r)
        out[(size_t)(row0 + r) * N_DIM + col] = acc[mt][nt][r] + bv;
    }
  }
}

extern "C" void kernel_launch(void* const* d_in, const int* in_sizes, int n_in,
                              void* d_out, int out_size, void* d_ws, size_t ws_size,
                              hipStream_t stream) {
  (void)n_in; (void)out_size;
  const float*        xp  = (const float*)d_in[0];
  const int*          qw  = (const int*)d_in[1];
  const unsigned int* qz  = (const unsigned int*)d_in[2];
  const float*        sc  = (const float*)d_in[3];
  const int*          gi  = (const int*)d_in[4];
  const float*        bi  = (const float*)d_in[5];
  float*              out = (float*)d_out;

  const int M = in_sizes[0] / K_DIM; // 8192
  const size_t need = (size_t)M * K_DIM * 2 + (size_t)N_DIM * K_DIM * 2; // 96MB

  dim3 grid(N_DIM / 128, M / 128);
  if (ws_size >= need && (M % 128) == 0) {
    unsigned short* xb = (unsigned short*)d_ws;
    unsigned short* wt = xb + (size_t)M * K_DIM;
    const int n8 = (M * K_DIM) / 8;
    cvt_x_bf16<<<(n8 + 255) / 256, 256, 0, stream>>>(xp, xb, n8);
    const int nw = (K_DIM / 8) * N_DIM;
    dequant_wt_bf16<<<nw / 256, 256, 0, stream>>>(qw, qz, sc, gi, wt);
    gemm_bf16_pipe<<<grid, 256, 0, stream>>>(xb, wt, bi, out);
  } else {
    gptq_gemm_fused<<<grid, 256, 0, stream>>>(xp, qw, qz, sc, gi, bi, out);
  }
}
